// GraphConvolution_52304111731093
// MI455X (gfx1250) — matrix operand order repaired and checked by reading, not compiled, not run on hardware
//
#include <hip/hip_runtime.h>

#define NN 50000
#define NE 1600000
#define DD 512

typedef __attribute__((ext_vector_type(16))) __bf16 v16bf;
typedef __attribute__((ext_vector_type(8)))  float  v8f;
typedef __attribute__((ext_vector_type(4)))  unsigned int v4u;

union FragBF { v4u q[2]; v16bf v; };   // 32 bytes: 16 bf16 elements per lane

__device__ __forceinline__ unsigned short f2bf(float f) {
  unsigned int u = __builtin_bit_cast(unsigned int, f);
  unsigned int r = 0x7FFFu + ((u >> 16) & 1u);   // round-to-nearest-even
  return (unsigned short)((u + r) >> 16);
}

__device__ __forceinline__ unsigned int pk2(float a, float b) {
  return (unsigned int)f2bf(a) | ((unsigned int)f2bf(b) << 16);
}

__device__ __forceinline__ v8f mm(const FragBF& a, const FragBF& b, v8f c) {
  return __builtin_amdgcn_wmma_f32_16x16x32_bf16(false, a.v, false, b.v,
                                                 (short)0, c, false, false);
}

// ---------------- zero workspace (y2 + colsum + colsumsq) ----------------
__global__ void k_zero(float4* __restrict__ p, int n4) {
  int i = blockIdx.x * 256 + threadIdx.x;
  if (i < n4) p[i] = make_float4(0.f, 0.f, 0.f, 0.f);
}

// ------- W fp32 [o][d] -> bf16 in per-lane B-fragment order ---------------
// Fragment (tile,s) = 1KB at (tile*16+s)*512 ushorts; lane L owns 16 ushorts
// at +L*16: K = s*32 + (L>>4)*8 + {0..7} and + {16..23} of column o = tile*16 + (L&15).
__global__ void k_swizw(const float* __restrict__ w,
                        unsigned short* __restrict__ wbfs) {
  int t = blockIdx.x * 256 + threadIdx.x;   // 16384 threads exactly
  int frag = t >> 5, L = t & 31;
  int tile = frag >> 4, s = frag & 15;
  int col = tile * 16 + (L & 15);
  int kb  = s * 32 + (L >> 4) * 8;
  const float* src = w + (size_t)col * DD + kb;
  v4u d0, d1;
  d0.x = pk2(src[0],  src[1]);   d0.y = pk2(src[2],  src[3]);
  d0.z = pk2(src[4],  src[5]);   d0.w = pk2(src[6],  src[7]);
  d1.x = pk2(src[16], src[17]);  d1.y = pk2(src[18], src[19]);
  d1.z = pk2(src[20], src[21]);  d1.w = pk2(src[22], src[23]);
  v4u* dst = (v4u*)(wbfs + (size_t)frag * 512 + L * 16);
  dst[0] = d0; dst[1] = d1;
}

// ------- x fp32 [n][d] -> bf16 in per-lane A-fragment order ---------------
// Fragment (rb,s) = 1KB at (rb*16+s)*512 ushorts; lane L owns 16 ushorts at
// +L*16: row = rb*16+(L&15), K = s*32 + aoff + {0..7} and + {16..23},
// aoff = (L>>4)*8  (ISA 16-bit A-matrix layout).
__global__ void k_swiza(const float* __restrict__ x,
                        unsigned short* __restrict__ xbfs) {
  int t = blockIdx.x * 256 + threadIdx.x;   // 1,600,000 threads exactly
  int frag = t >> 5, L = t & 31;
  int rb = frag >> 4, s = frag & 15;
  int row  = rb * 16 + (L & 15);
  int aoff = (L >> 4) * 8;
  const float* src = x + (size_t)row * DD + s * 32 + aoff;
  v4u d0, d1;
  d0.x = pk2(src[0],  src[1]);   d0.y = pk2(src[2],  src[3]);
  d0.z = pk2(src[4],  src[5]);   d0.w = pk2(src[6],  src[7]);
  d1.x = pk2(src[16], src[17]);  d1.y = pk2(src[18], src[19]);
  d1.z = pk2(src[20], src[21]);  d1.w = pk2(src[22], src[23]);
  v4u* dst = (v4u*)(xbfs + (size_t)frag * 512 + L * 16);
  dst[0] = d0; dst[1] = d1;
}

// ---------------- GEMM: y = x @ W^T (bf16 WMMA, swizzled operands) --------
// block = 256 threads (8 waves); block owns 32 rows x 512 cols.
// Wave wv: 2 M-tiles (row-blocks 2b, 2b+1) x 4 N-tiles (cols wv*64..+64).
// All fragment loads are contiguous 2x16B per lane in pre-swizzled layout.
__global__ void __launch_bounds__(256) k_gemm(const unsigned short* __restrict__ xbfs,
                                              const unsigned short* __restrict__ wbfs,
                                              float* __restrict__ y) {
  const int lane = threadIdx.x & 31;
  const int wv   = threadIdx.x >> 5;            // 0..7
  const int rb1  = blockIdx.x * 2;
  const bool valid2 = (rb1 + 1) < 3125;         // M raggedness (3125 row-blocks)
  const int rb2  = valid2 ? (rb1 + 1) : 3124;   // clamp loads; stores guarded

  const unsigned short* ap1 = xbfs + (size_t)rb1 * 16 * 512 + lane * 16;
  const unsigned short* ap2 = xbfs + (size_t)rb2 * 16 * 512 + lane * 16;
  const unsigned short* bp  = wbfs + (size_t)(wv * 4) * 16 * 512 + lane * 16;

  v8f acc[2][4] = {};

#pragma unroll
  for (int s = 0; s < 16; ++s) {                // K slices of 32
    FragBF a1, a2, b[4];
    a1.q[0] = *(const v4u*)(ap1 + s * 512);
    a1.q[1] = *(const v4u*)(ap1 + s * 512 + 8);
    a2.q[0] = *(const v4u*)(ap2 + s * 512);
    a2.q[1] = *(const v4u*)(ap2 + s * 512 + 8);
#pragma unroll
    for (int n = 0; n < 4; ++n) {
      b[n].q[0] = *(const v4u*)(bp + (n * 16 + s) * 512);
      b[n].q[1] = *(const v4u*)(bp + (n * 16 + s) * 512 + 8);
    }
#pragma unroll
    for (int n = 0; n < 4; ++n) {               // each b feeds 2 WMMAs
      acc[0][n] = mm(a1, b[n], acc[0][n]);
      acc[1][n] = mm(a2, b[n], acc[1][n]);
    }
  }

  // C layout: VGPR r -> M = r + 8*(lane>=16), N = lane&15
  const int mrow = lane & 15;
  const int aoff = (lane >> 4) * 8;
  const int colb = wv * 64;
#pragma unroll
  for (int n = 0; n < 4; ++n)
#pragma unroll
    for (int r = 0; r < 8; ++r)
      y[(size_t)(rb1 * 16 + aoff + r) * DD + colb + n * 16 + mrow] = acc[0][n][r];
  if (valid2) {
#pragma unroll
    for (int n = 0; n < 4; ++n)
#pragma unroll
      for (int r = 0; r < 8; ++r)
        y[(size_t)(rb2 * 16 + aoff + r) * DD + colb + n * 16 + mrow] = acc[1][n][r];
  }
}

// ---------------- SpMM: y2[row,:] += val * y[col,:]  (1 wave / edge) ------
__global__ void __launch_bounds__(256) k_spmm(const float* __restrict__ y,
                                              const int* __restrict__ rows,
                                              const int* __restrict__ cols,
                                              const float* __restrict__ vals,
                                              float* __restrict__ y2) {
  const int lane = threadIdx.x & 31;
  const int e = blockIdx.x * 8 + (threadIdx.x >> 5);   // NE/8 blocks exactly
  const int r = rows[e];
  const int c = cols[e];
  const float v = vals[e];
  const float* src = y  + (size_t)c * DD;
  float*       dst = y2 + (size_t)r * DD;
#pragma unroll
  for (int i = 0; i < 16; ++i) {
    unsafeAtomicAdd(&dst[lane + i * 32], v * src[lane + i * 32]);
  }
}

// ---------------- per-column sum / sumsq partials -------------------------
__global__ void __launch_bounds__(256) k_colstats(const float* __restrict__ y2,
                                                  float* __restrict__ colsum,
                                                  float* __restrict__ colsumsq) {
  const int d = threadIdx.x;                  // columns d and d+256
  float s1a = 0.f, s2a = 0.f, s1b = 0.f, s2b = 0.f;
  for (int r = blockIdx.x; r < NN; r += gridDim.x) {
    float a = y2[(size_t)r * DD + d];
    float b = y2[(size_t)r * DD + d + 256];
    s1a += a; s2a += a * a;
    s1b += b; s2b += b * b;
  }
  unsafeAtomicAdd(&colsum[d], s1a);
  unsafeAtomicAdd(&colsumsq[d], s2a);
  unsafeAtomicAdd(&colsum[d + 256], s1b);
  unsafeAtomicAdd(&colsumsq[d + 256], s2b);
}

// ---------------- finalize mean + global RMS coefficient ------------------
__global__ void __launch_bounds__(512) k_stats(const float* __restrict__ colsum,
                                               const float* __restrict__ colsumsq,
                                               const float* __restrict__ scale,
                                               float* __restrict__ mean,
                                               float* __restrict__ coef) {
  __shared__ float sh[512];
  const int d = threadIdx.x;
  const float s = colsum[d];
  const float m = s * (1.0f / NN);
  mean[d] = m;
  sh[d] = colsumsq[d] - s * m;   // = sum_n (y-m)^2 for column d
  __syncthreads();
  for (int off = 256; off > 0; off >>= 1) {
    if (d < off) sh[d] += sh[d + off];
    __syncthreads();
  }
  if (d == 0) {
    float msq = sh[0] * (1.0f / NN);   // mean over nodes of ||y_t||^2
    coef[0] = rsqrtf(msq) * (1.0f + scale[0]) * 22.62741699796952f; // sqrt(512)
  }
}

// ---------------- fused center/scale + relu + residual --------------------
__global__ void __launch_bounds__(256) k_final(const float* __restrict__ y2,
                                               const float* __restrict__ x,
                                               const float* __restrict__ mean,
                                               const float* __restrict__ coef,
                                               float* __restrict__ out) {
  const size_t i4 = (size_t)blockIdx.x * 256 + threadIdx.x;  // float4 index
  const float c = coef[0];
  const float4 v  = ((const float4*)y2)[i4];
  const float4 xv = ((const float4*)x)[i4];
  const float4 mv = ((const float4*)mean)[i4 & 127];         // (i4*4) % 512
  float4 o;
  o.x = fmaxf((v.x - mv.x) * c, 0.f) + xv.x;
  o.y = fmaxf((v.y - mv.y) * c, 0.f) + xv.y;
  o.z = fmaxf((v.z - mv.z) * c, 0.f) + xv.z;
  o.w = fmaxf((v.w - mv.w) * c, 0.f) + xv.w;
  ((float4*)out)[i4] = o;
}

extern "C" void kernel_launch(void* const* d_in, const int* in_sizes, int n_in,
                              void* d_out, int out_size, void* d_ws, size_t ws_size,
                              hipStream_t stream) {
  const float* x     = (const float*)d_in[0];
  const int*   rows  = (const int*)d_in[1];
  const int*   cols  = (const int*)d_in[2];
  const float* vals  = (const float*)d_in[3];
  const float* w     = (const float*)d_in[4];
  const float* scale = (const float*)d_in[5];
  float* out = (float*)d_out;

  // workspace layout (float indices)
  float* wsf      = (float*)d_ws;
  float* y2       = wsf;                 // 25,600,000 floats
  float* colsum   = wsf + 25600000;      // 512
  float* colsumsq = wsf + 25600512;      // 512
  float* mean     = wsf + 25601024;      // 512
  float* coef     = wsf + 25601536;      // 1
  unsigned short* wbfs = (unsigned short*)(wsf + 25601792);  // 512*512 bf16 (swizzled)
  unsigned short* xbfs = (unsigned short*)(wsf + 25732864);  // 50000*512 bf16 (swizzled)

  // zero y2 + colsum + colsumsq = 25,601,024 floats = 6,400,256 float4
  k_zero<<<25001, 256, 0, stream>>>((float4*)wsf, 6400256);
  k_swizw<<<64, 256, 0, stream>>>(w, wbfs);                  // 512*512/16 threads
  k_swiza<<<6250, 256, 0, stream>>>(x, xbfs);                // 50000*512/16 threads
  k_gemm<<<1563, 256, 0, stream>>>(xbfs, wbfs, out);         // y -> d_out (reused)
  k_spmm<<<NE / 8, 256, 0, stream>>>(out, rows, cols, vals, y2);
  k_colstats<<<512, 256, 0, stream>>>(y2, colsum, colsumsq);
  k_stats<<<1, 512, 0, stream>>>(colsum, colsumsq, scale, mean, coef);
  k_final<<<25000, 256, 0, stream>>>(y2, x, mean, coef, out);
}
